// SimpleSSM_31911607009507
// MI455X (gfx1250) — matrix operand order repaired and checked by reading, not compiled, not run on hardware
//
#include <hip/hip_runtime.h>
#include <hip/hip_bf16.h>
#include <math.h>
#include <stdint.h>

#define VOCAB 32000
#define EMB   512
#define STATE 512
#define BB    4
#define SS    1024
#define NWG_SCAN 32

typedef __attribute__((ext_vector_type(2)))  float  v2f;
typedef __attribute__((ext_vector_type(4)))  float  f32x4;
typedef __attribute__((ext_vector_type(8)))  float  v8f;
typedef __attribute__((ext_vector_type(4)))  int    v4i;
typedef __attribute__((ext_vector_type(8)))  __bf16 bf16x8;
typedef __attribute__((ext_vector_type(16))) __bf16 bf16x16;

union BF16Frag { bf16x16 v; bf16x8 h[2]; };

// Typed address-space-qualified pointers for the async global->LDS builtin:
// param0 = v4i addrspace(1)*, param1 = v4i addrspace(3)*.
typedef __attribute__((address_space(1))) v4i* gv4i_t;
typedef __attribute__((address_space(3))) v4i* lv4i_t;
#define AS1PTR(p) ((gv4i_t)(uintptr_t)(p))
#define AS3PTR(p) ((lv4i_t)(unsigned int)(uintptr_t)(p))

// ---------------------------------------------------------------------------
// Prep kernels: fp32 -> bf16 conversions, b transpose, state/counter init.
// ---------------------------------------------------------------------------
__global__ void prep_cw_kernel(const float* __restrict__ cw, __bf16* __restrict__ cwb) {
    size_t n = (size_t)VOCAB * STATE;
    for (size_t i = (size_t)blockIdx.x * blockDim.x + threadIdx.x; i < n;
         i += (size_t)gridDim.x * blockDim.x)
        cwb[i] = (__bf16)cw[i];
}

__global__ void prep_bT_kernel(const float* __restrict__ bmat, __bf16* __restrict__ bT) {
    int i = blockIdx.x * blockDim.x + threadIdx.x;
    if (i < EMB * STATE) {
        int e = i >> 9;        // row of b  (EMB)
        int nn = i & 511;      // col of b  (STATE)
        bT[nn * EMB + e] = (__bf16)bmat[i];   // bT[n][e] = b[e][n]
    }
}

__global__ void prep_state_kernel(float* st0, float* st1, unsigned* cnt) {
    int i = blockIdx.x * blockDim.x + threadIdx.x;
    if (i < 16 * STATE) { st0[i] = 0.0f; st1[i] = 0.0f; }
    if (i == 0) *cnt = 0u;
}

// ---------------------------------------------------------------------------
// Kernel 1: u[b*S+s, n] = sum_e emb_table[x[b*S+s], e] * b[e, n]
// bf16 WMMA 16x16x32, f32 accumulate. Block tile 128x128 (8 waves, 64x32 each).
// ---------------------------------------------------------------------------
__global__ __launch_bounds__(256) void uproj_kernel(
    const int*    __restrict__ x,     // [B*S]
    const float*  __restrict__ embt,  // [VOCAB, EMB] fp32
    const __bf16* __restrict__ bT,    // [STATE(n), EMB(e)] bf16 (B col-major)
    float*        __restrict__ u)     // [B*S, STATE] fp32
{
    const int tid  = threadIdx.x;
    const int lane = tid & 31;
    const int wid  = tid >> 5;
    const int wy   = wid >> 2;            // 0..1  -> M
    const int wx   = wid & 3;             // 0..3  -> N
    const int l15  = lane & 15;
    const int h    = lane >> 4;

    const int m_w = blockIdx.y * 128 + wy * 64;   // 4 M-subtiles of 16
    const int n_w = blockIdx.x * 128 + wx * 32;   // 2 N-subtiles of 16

    int tok[4];
#pragma unroll
    for (int mi = 0; mi < 4; ++mi) tok[mi] = x[m_w + mi * 16 + l15];

    v8f acc[4][2];
#pragma unroll
    for (int mi = 0; mi < 4; ++mi)
#pragma unroll
        for (int ni = 0; ni < 2; ++ni) acc[mi][ni] = (v8f)(0.0f);

    for (int k0 = 0; k0 < EMB; k0 += 32) {
        BF16Frag af[4], bfr[2];
#pragma unroll
        for (int mi = 0; mi < 4; ++mi) {
            const float* ep = embt + (size_t)tok[mi] * EMB + k0 + 8 * h;
            f32x4 w0 = *(const f32x4*)(ep);
            f32x4 w1 = *(const f32x4*)(ep + 4);
            f32x4 w2 = *(const f32x4*)(ep + 16);
            f32x4 w3 = *(const f32x4*)(ep + 20);
#pragma unroll
            for (int j = 0; j < 4; ++j) {
                af[mi].v[j]      = (__bf16)w0[j];
                af[mi].v[4 + j]  = (__bf16)w1[j];
                af[mi].v[8 + j]  = (__bf16)w2[j];
                af[mi].v[12 + j] = (__bf16)w3[j];
            }
        }
#pragma unroll
        for (int ni = 0; ni < 2; ++ni) {
            const __bf16* bp = bT + (size_t)(n_w + ni * 16 + l15) * EMB + k0 + 8 * h;
            bfr[ni].h[0] = *(const bf16x8*)(bp);
            bfr[ni].h[1] = *(const bf16x8*)(bp + 16);
        }
#pragma unroll
        for (int mi = 0; mi < 4; ++mi)
#pragma unroll
            for (int ni = 0; ni < 2; ++ni)
                acc[mi][ni] = __builtin_amdgcn_wmma_f32_16x16x32_bf16(
                    false, af[mi].v, false, bfr[ni].v, (short)0, acc[mi][ni],
                    false, false);
    }

#pragma unroll
    for (int mi = 0; mi < 4; ++mi)
#pragma unroll
        for (int ni = 0; ni < 2; ++ni) {
            int col = n_w + ni * 16 + l15;
#pragma unroll
            for (int e = 0; e < 8; ++e) {
                int r = m_w + mi * 16 + e + 8 * h;
                u[(size_t)r * STATE + col] = acc[mi][ni][e];
            }
        }
}

// ---------------------------------------------------------------------------
// Kernel 2: sequential scan. 32 persistent WGs; WG w owns state columns
// [16w, 16w+16). a-slice cached in LDS fp32 (padded to kill bank conflicts).
// state = tanh(state @ a + u_s) in fp32 via v_wmma_f32_16x16x4_f32, 4-wave
// K-split + LDS reduction, cross-WG sync via monotonic atomic barrier.
// State padded to 16 rows (rows 4..15 stay zero) so WMMA tiles are clean.
// ---------------------------------------------------------------------------
__global__ __launch_bounds__(128) void scan_kernel(
    const float* __restrict__ amat,     // [STATE, STATE] fp32 (a[k][n])
    const float* __restrict__ u,        // [B*S, STATE] fp32
    float* st0, float* st1,             // [16, STATE] ping-pong (zero-init)
    __bf16* __restrict__ states_bf,     // [S*B, STATE] bf16 out
    unsigned* __restrict__ cnt)         // barrier counter (zero-init)
{
    __shared__ float alds[512 * 17];    // 512 K-rows x 16 cols, pad 17
    __shared__ float red[4][32][8];     // per-wave WMMA partials

    const int tid  = threadIdx.x;       // 128
    const int lane = tid & 31;
    const int wv   = tid >> 5;          // 0..3 K-split
    const int n0   = blockIdx.x * 16;   // this WG's column base
    const int l15  = lane & 15;
    const int h    = lane >> 4;

    // Stage a[:, n0:n0+16] into LDS.
    for (int i = tid; i < 512 * 16; i += 128) {
        int k = i >> 4, c = i & 15;
        alds[k * 17 + c] = amat[(size_t)k * STATE + n0 + c];
    }
    __syncthreads();

    float* st[2] = { st0, st1 };

    for (int s = 0; s < SS; ++s) {
        const volatile float* cur = st[s & 1];
        float* nxt = st[(s + 1) & 1];

        // Partial [16x16] tile over K in [wv*128, wv*128+128).
        v8f acc = (v8f)(0.0f);
        const int row = l15;            // state row (0..15, rows>=4 are zero)
#pragma unroll 4
        for (int ks = 0; ks < 32; ++ks) {
            int k0 = wv * 128 + ks * 4;
            v2f afr, bfr;
            afr[0] = cur[row * STATE + k0 + 2 * h];
            afr[1] = cur[row * STATE + k0 + 2 * h + 1];
            bfr[0] = alds[(k0 + 2 * h) * 17 + l15];
            bfr[1] = alds[(k0 + 2 * h + 1) * 17 + l15];
            acc = __builtin_amdgcn_wmma_f32_16x16x4_f32(
                false, afr, false, bfr, (short)0, acc, false, false);
        }
#pragma unroll
        for (int e = 0; e < 8; ++e) red[wv][lane][e] = acc[e];
        __syncthreads();

        // Wave 0 reduces 4 partials, adds u, applies tanh, writes rows 0..3.
        if (wv == 0 && h == 0) {
#pragma unroll
            for (int e = 0; e < 4; ++e) {          // e == batch index b, row e
                float sum = red[0][lane][e] + red[1][lane][e] +
                            red[2][lane][e] + red[3][lane][e];
                float uu = u[((size_t)e * SS + s) * STATE + n0 + l15];
                float v  = tanhf(sum + uu);
                nxt[e * STATE + n0 + l15] = v;
                states_bf[((size_t)s * BB + e) * STATE + n0 + l15] = (__bf16)v;
            }
        }
        __threadfence();
        __syncthreads();
        if (tid == 0) {
            atomicAdd(cnt, 1u);
            unsigned target = (unsigned)(NWG_SCAN * (s + 1));
            while (__hip_atomic_load(cnt, __ATOMIC_ACQUIRE,
                                     __HIP_MEMORY_SCOPE_AGENT) < target)
                __builtin_amdgcn_s_sleep(2);
        }
        __syncthreads();
        __threadfence();
    }
}

// ---------------------------------------------------------------------------
// Kernel 3: logits[b,s,v] = states[s*B+b,:] . c_w[v,:] + c_b[v]
// bf16 WMMA 16x16x32, f32 accumulate. Block tile 128x128. A/B tiles are
// staged per-block into LDS with gfx1250 async global->LDS DMA (ASYNCcnt),
// double-buffered over K-chunks of 32, removing the 4x/2x intra-block L2
// fragment redundancy. HBM traffic ~= the 524 MB fp32 output stream.
// ---------------------------------------------------------------------------
#define KCHUNK   32
#define NKCHUNK  (STATE / KCHUNK)      // 16
#define LDS_ROWP 40                    // 32 K elems padded to 40 (bank spread)

__global__ __launch_bounds__(256) void readout_kernel(
    const __bf16* __restrict__ Abf,    // [S*B, STATE] states bf16
    const __bf16* __restrict__ Wbf,    // [VOCAB, STATE] c_w bf16
    const float*  __restrict__ bias,   // [VOCAB] fp32
    float*        __restrict__ out)    // [B, S, VOCAB] fp32
{
    __shared__ __align__(16) __bf16 Alds[2][128 * LDS_ROWP];
    __shared__ __align__(16) __bf16 Blds[2][128 * LDS_ROWP];

    const int tid  = threadIdx.x;
    const int lane = tid & 31;
    const int wid  = tid >> 5;
    const int wy   = wid >> 2;
    const int wx   = wid & 3;
    const int l15  = lane & 15;
    const int h    = lane >> 4;

    const int m_blk = blockIdx.y * 128;
    const int n_blk = blockIdx.x * 128;

    // Issue one 32-wide K-chunk of the block's A and B tiles as async DMA.
    // 128 rows x 32 K x bf16 = 512 b128 units per tile; 2 A + 2 B per thread
    // = 4 async instructions per wave per chunk (in-order completion).
    auto issue = [&](int kc) {
        int buf = kc & 1;
#pragma unroll
        for (int j = 0; j < 2; ++j) {
            int unit = tid + 256 * j;          // 0..511
            int row  = unit >> 2;              // 0..127
            int seg  = unit & 3;               // 8-elem segment within chunk
            const __bf16* ga = Abf + (size_t)(m_blk + row) * STATE + kc * KCHUNK + seg * 8;
            __builtin_amdgcn_global_load_async_to_lds_b128(
                AS1PTR(ga), AS3PTR(&Alds[buf][row * LDS_ROWP + seg * 8]), 0, 0);
            const __bf16* gb = Wbf + (size_t)(n_blk + row) * STATE + kc * KCHUNK + seg * 8;
            __builtin_amdgcn_global_load_async_to_lds_b128(
                AS1PTR(gb), AS3PTR(&Blds[buf][row * LDS_ROWP + seg * 8]), 0, 0);
        }
    };

    v8f acc[4][2];
#pragma unroll
    for (int mi = 0; mi < 4; ++mi)
#pragma unroll
        for (int ni = 0; ni < 2; ++ni) acc[mi][ni] = (v8f)(0.0f);

    issue(0);
    for (int kc = 0; kc < NKCHUNK; ++kc) {
        const int buf = kc & 1;
        if (kc + 1 < NKCHUNK) {
            issue(kc + 1);
            __builtin_amdgcn_s_wait_asynccnt(4);  // chunk kc landed (in-order)
        } else {
            __builtin_amdgcn_s_wait_asynccnt(0);
        }
        __syncthreads();                           // all waves' DMA visible

        BF16Frag af[4], bfr[2];
#pragma unroll
        for (int mi = 0; mi < 4; ++mi) {
            const __bf16* ap = &Alds[buf][(wy * 64 + mi * 16 + l15) * LDS_ROWP + 8 * h];
            af[mi].h[0] = *(const bf16x8*)(ap);
            af[mi].h[1] = *(const bf16x8*)(ap + 16);
        }
#pragma unroll
        for (int ni = 0; ni < 2; ++ni) {
            const __bf16* bp = &Blds[buf][(wx * 32 + ni * 16 + l15) * LDS_ROWP + 8 * h];
            bfr[ni].h[0] = *(const bf16x8*)(bp);
            bfr[ni].h[1] = *(const bf16x8*)(bp + 16);
        }
#pragma unroll
        for (int mi = 0; mi < 4; ++mi)
#pragma unroll
            for (int ni = 0; ni < 2; ++ni)
                acc[mi][ni] = __builtin_amdgcn_wmma_f32_16x16x32_bf16(
                    false, af[mi].v, false, bfr[ni].v, (short)0, acc[mi][ni],
                    false, false);
        __syncthreads();                           // done reading before reuse
    }

#pragma unroll
    for (int mi = 0; mi < 4; ++mi)
#pragma unroll
        for (int ni = 0; ni < 2; ++ni) {
            int col  = n_blk + wx * 32 + ni * 16 + l15;
            float bv = bias[col];
#pragma unroll
            for (int e = 0; e < 8; ++e) {
                int r = m_blk + wy * 64 + mi * 16 + e + 8 * h;   // r = s*B + b
                int b = r & 3;
                int sidx = r >> 2;
                out[(size_t)b * SS * VOCAB + (size_t)sidx * VOCAB + col] =
                    acc[mi][ni][e] + bv;
            }
        }
}

// ---------------------------------------------------------------------------
extern "C" void kernel_launch(void* const* d_in, const int* in_sizes, int n_in,
                              void* d_out, int out_size, void* d_ws, size_t ws_size,
                              hipStream_t stream) {
    const int*   x    = (const int*)d_in[0];
    const float* embt = (const float*)d_in[1];
    const float* amat = (const float*)d_in[2];
    const float* bmat = (const float*)d_in[3];
    const float* cw   = (const float*)d_in[4];
    const float* cb   = (const float*)d_in[5];
    float* out = (float*)d_out;

    char* ws = (char*)d_ws;
    size_t off = 0;
    float* u = (float*)(ws + off);          off += (size_t)BB * SS * STATE * 4;  // 8 MB
    float* st0 = (float*)(ws + off);        off += (size_t)16 * STATE * 4;
    float* st1 = (float*)(ws + off);        off += (size_t)16 * STATE * 4;
    unsigned* cnt = (unsigned*)(ws + off);  off += 256;
    __bf16* states_bf = (__bf16*)(ws + off); off += (size_t)SS * BB * STATE * 2; // 4 MB
    __bf16* bT = (__bf16*)(ws + off);       off += (size_t)EMB * STATE * 2;      // 0.5 MB
    __bf16* cwb = (__bf16*)(ws + off);      off += (size_t)VOCAB * STATE * 2;    // 32.8 MB

    prep_cw_kernel<<<4096, 256, 0, stream>>>(cw, cwb);
    prep_bT_kernel<<<(EMB * STATE + 255) / 256, 256, 0, stream>>>(bmat, bT);
    prep_state_kernel<<<(16 * STATE + 255) / 256, 256, 0, stream>>>(st0, st1, cnt);

    uproj_kernel<<<dim3(STATE / 128, (BB * SS) / 128), 256, 0, stream>>>(x, embt, bT, u);
    scan_kernel<<<NWG_SCAN, 128, 0, stream>>>(amat, u, st0, st1, states_bf, cnt);
    readout_kernel<<<dim3(VOCAB / 128, (BB * SS) / 128), 256, 0, stream>>>(
        states_bf, cwb, cb, out);
}
